// fconv1d_19129784337064
// MI455X (gfx1250) — matrix operand order repaired and checked by reading, not compiled, not run on hardware
//
#include <hip/hip_runtime.h>

// CDNA5 / gfx1250 implicit-GEMM conv1d:
//   z[n, a] = sum_{b,t} x[n+t, b] * kernel[256-t, a, b]   (+ bias[a])
// M=16128 rows, N=64 output channels, K=257*64=16448 reduction.
// f16 WMMA operands, f32 accumulation. Weights staged to LDS per-t via
// async global->LDS copies (double buffered), shared by 4 waves/block.

typedef __attribute__((ext_vector_type(16))) _Float16 v16h;
typedef __attribute__((ext_vector_type(8)))  _Float16 v8h;
typedef __attribute__((ext_vector_type(8)))  float    v8f;
typedef __attribute__((ext_vector_type(4)))  int      v4i;

typedef __attribute__((address_space(1))) v4i g_v4i;  // global
typedef __attribute__((address_space(3))) v4i l_v4i;  // LDS

#define N_IN    16384
#define CI_     64
#define CO_     64
#define KW_     257
#define N_OUT_  (N_IN - KW_ + 1)      // 16128

#define XH_HALFS   (N_IN * CI_)       // 1048576 (2 MB f16)
#define WH_HALFS   (KW_ * CI_ * CO_)  // 1052672 (~2.06 MB f16)
#define SLAB_HALFS (CI_ * CO_)        // 4096 halfs = 8 KB per t

// ---- async global->LDS path (guarded; falls back to plain copies) --------
#if defined(__HIP_DEVICE_COMPILE__) && __has_builtin(__builtin_amdgcn_global_load_async_to_lds_b128)
#define USE_ASYNC 1
#else
#define USE_ASYNC 0
#endif

#if USE_ASYNC
#if __has_builtin(__builtin_amdgcn_s_wait_asynccnt)
#define WAIT_ASYNC(n) __builtin_amdgcn_s_wait_asynccnt(n)
#else
#define WAIT_ASYNC(n) asm volatile("s_wait_asynccnt %0" ::"i"(n) : "memory")
#endif
#else
#define WAIT_ASYNC(n) ((void)0)
#endif

// Copy one 8 KB weight slab (4096 halfs) to LDS with 128 threads:
// 4 chunks of 16 B per thread. Layout-preserving (fragment order kept).
__device__ __forceinline__ void stage_slab(const _Float16* __restrict__ src,
                                           _Float16* dst, int tid) {
#if USE_ASYNC
#pragma unroll
  for (int j = 0; j < 4; ++j) {
    int off = tid * 8 + j * 1024;
    __builtin_amdgcn_global_load_async_to_lds_b128(
        (g_v4i*)(src + off), (l_v4i*)(dst + off), 0, 0);
  }
#else
#pragma unroll
  for (int j = 0; j < 4; ++j) {
    int off = tid * 8 + j * 1024;
    *(v8h*)(dst + off) = *(const v8h*)(src + off);
  }
#endif
}

// ---- prep kernels --------------------------------------------------------

__global__ __launch_bounds__(256) void k_cvt_x(const float* __restrict__ x,
                                               _Float16* __restrict__ xh) {
  int i = blockIdx.x * 256 + threadIdx.x;
  if (i < XH_HALFS) xh[i] = (_Float16)x[i];
}

// Pack flipped kernel into B-fragment order:
// wh[((t*2 + kc)*4 + nt)*512 + lane*16 + h] =
//   kernel[256 - t][co = nt*16 + (lane&15)][ci = kc*32 + (lane>>4)*16 + h]
// (B 32x16 f16: lane holds one column; lanes 0-15 K=0..15, lanes 16-31 K=16..31)
__global__ __launch_bounds__(256) void k_pack_w(const float* __restrict__ w,
                                                _Float16* __restrict__ wh) {
  int i = blockIdx.x * 256 + threadIdx.x;
  if (i >= WH_HALFS) return;
  int h    = i & 15;
  int lane = (i >> 4) & 31;
  int nt   = (i >> 9) & 3;
  int kc   = (i >> 11) & 1;
  int t    = i >> 12;
  int co = nt * 16 + (lane & 15);
  int ci = kc * 32 + (lane >> 4) * 16 + h;
  wh[i] = (_Float16)w[((size_t)(KW_ - 1 - t) * CO_ + co) * CI_ + ci];
}

__global__ __launch_bounds__(256) void k_zero(float* __restrict__ out, int n) {
  int i = blockIdx.x * 256 + threadIdx.x;
  if (i < n) out[i] = 0.0f;
}

// ---- main WMMA kernel ----------------------------------------------------
// grid = (126, 2), block = 128 (4 waves). Wave tile: 32 rows x 64 co.
// blockIdx.y splits the t range (129 + 128) -> exactly 2 atomic adds per
// output element on a zeroed buffer => bitwise deterministic.

union Frag {
  v16h v;
  v8h  h[2];
};

__global__ __launch_bounds__(128) void k_conv_wmma(
    const _Float16* __restrict__ xh, const _Float16* __restrict__ wh,
    const float* __restrict__ bias, float* __restrict__ out) {
  __shared__ _Float16 lds[2][SLAB_HALFS];  // 16 KB double buffer

  const int tid  = threadIdx.x;
  const int wave = tid >> 5;
  const int lane = tid & 31;
  const int rlo  = lane & 15;  // row-in-tile (A) / col-in-tile (B,C)
  const int khi  = lane >> 4;  // K-half select (A) / M-half select (C)
  const int n0   = blockIdx.x * 128 + wave * 32;

  const int tBeg = (blockIdx.y == 0) ? 0 : 129;
  const int tEnd = (blockIdx.y == 0) ? 129 : KW_;

  v8f zero = {};
  v8f acc[2][4];
#pragma unroll
  for (int mt = 0; mt < 2; ++mt)
#pragma unroll
    for (int nt = 0; nt < 4; ++nt) acc[mt][nt] = zero;

  stage_slab(wh + (size_t)tBeg * SLAB_HALFS, &lds[tBeg & 1][0], tid);

  for (int t = tBeg; t < tEnd; ++t) {
    if (t + 1 < tEnd) {
      stage_slab(wh + (size_t)(t + 1) * SLAB_HALFS, &lds[(t + 1) & 1][0], tid);
      WAIT_ASYNC(4);  // slab t done; slab t+1 (4 newest) may be in flight
    } else {
      WAIT_ASYNC(0);
    }
    __syncthreads();  // all waves' portions of slab t visible

    const _Float16* sl = &lds[t & 1][0];
#pragma unroll
    for (int kc = 0; kc < 2; ++kc) {
      // A fragments: 16x32 f16. Lane holds row rlo; halfs cover
      // K = kc*32 + khi*16 .. kc*32 + khi*16 + 15 (contiguous, matches B pack).
      Frag a[2];
#pragma unroll
      for (int mt = 0; mt < 2; ++mt) {
        const _Float16* ap =
            xh + (size_t)(n0 + mt * 16 + rlo + t) * CI_ + kc * 32 + khi * 16;
        a[mt].h[0] = *(const v8h*)ap;
        a[mt].h[1] = *(const v8h*)(ap + 8);
      }
#pragma unroll
      for (int nt = 0; nt < 4; ++nt) {
        Frag b;
        const _Float16* bp = sl + (kc * 4 + nt) * 512 + lane * 16;
        b.h[0] = *(const v8h*)bp;
        b.h[1] = *(const v8h*)(bp + 8);
#pragma unroll
        for (int mt = 0; mt < 2; ++mt) {
          acc[mt][nt] = __builtin_amdgcn_wmma_f32_16x16x32_f16(
              false, a[mt].v, false, b.v, (short)0, acc[mt][nt], false, false);
        }
      }
    }
    __syncthreads();  // compute(t) done before slab t's buffer is re-staged
  }

  // Epilogue: C/D 16x16 f32 layout: lane = col (rlo), VGPR v = row khi*8+v.
  // Two splits -> two commutative f32 atomic adds per element (deterministic).
#pragma unroll
  for (int nt = 0; nt < 4; ++nt) {
    float badd = (blockIdx.y == 0) ? bias[nt * 16 + rlo] : 0.0f;
#pragma unroll
    for (int mt = 0; mt < 2; ++mt) {
#pragma unroll
      for (int v = 0; v < 8; ++v) {
        int row = n0 + mt * 16 + khi * 8 + v;
        atomicAdd(&out[(size_t)row * CO_ + nt * 16 + rlo],
                  acc[mt][nt][v] + badd);
      }
    }
  }
}

// ---- launcher ------------------------------------------------------------

extern "C" void kernel_launch(void* const* d_in, const int* in_sizes, int n_in,
                              void* d_out, int out_size, void* d_ws,
                              size_t ws_size, hipStream_t stream) {
  const float* x    = (const float*)d_in[0];  // (16384, 64)
  const float* w    = (const float*)d_in[1];  // (257, 64, 64)
  const float* bias = (const float*)d_in[2];  // (64,)
  float* out = (float*)d_out;                 // (16128, 64)

  _Float16* xh = (_Float16*)d_ws;  // 2 MB
  _Float16* wh = (_Float16*)((char*)d_ws + (size_t)XH_HALFS * sizeof(_Float16));
  // total workspace use: ~4.2 MB

  k_cvt_x<<<(XH_HALFS + 255) / 256, 256, 0, stream>>>(x, xh);
  k_pack_w<<<(WH_HALFS + 255) / 256, 256, 0, stream>>>(w, wh);
  k_zero<<<(out_size + 255) / 256, 256, 0, stream>>>(out, out_size);

  dim3 grid(N_OUT_ / 128, 2);  // 126 row-blocks x 2 t-splits = 252 blocks
  k_conv_wmma<<<grid, 128, 0, stream>>>(xh, wh, bias, out);
}
